// lstm_RN_12472585028155
// MI455X (gfx1250) — matrix operand order repaired and checked by reading, not compiled, not run on hardware
//
#include <hip/hip_runtime.h>

typedef _Float16 f16;
typedef __attribute__((ext_vector_type(16))) _Float16 v16h;
typedef __attribute__((ext_vector_type(8)))  _Float16 v8h;
typedef __attribute__((ext_vector_type(8)))  float    v8f;

#define B_   256
#define T_   64
#define V_   1000
#define O_   32
#define H_   128
#define G4H  512

// ---------------------------------------------------------------------------
// Weight convert: src f32 [N][K] row-major -> dst f16 [Np][Kp] row-major,
// zero padded. NOTE: [n][k] row-major matches the B-fragment order this code uses
// (lane = n row, contiguous k, same k order as A), so B loads straight from
// global with no LDS transpose.
// ---------------------------------------------------------------------------
__global__ __launch_bounds__(256) void cvt_weight16_k(
    const float* __restrict__ src, f16* __restrict__ dst,
    int N, int K, int Np, int Kp)
{
    int idx = blockIdx.x * 256 + threadIdx.x;
    if (idx >= Np * Kp) return;
    int n = idx / Kp, k = idx % Kp;
    float v = (n < N && k < K) ? src[(size_t)n * K + k] : 0.0f;
    dst[idx] = (f16)v;
}

// Activation convert: src f32 [M][K] -> dst f16 [M][Kp], zero padded.
__global__ __launch_bounds__(256) void cvt_act16_k(
    const float* __restrict__ src, f16* __restrict__ dst, int M, int K, int Kp)
{
    int idx = blockIdx.x * 256 + threadIdx.x;
    if (idx >= M * Kp) return;
    int m = idx / Kp, k = idx % Kp;
    dst[idx] = (f16)((k < K) ? src[(size_t)m * K + k] : 0.0f);
}

// Combined LSTM bias + initial h (f16) / c (f32) state.
__global__ __launch_bounds__(256) void init_lstm_k(
    const float* __restrict__ b_ih, const float* __restrict__ b_hh,
    float* __restrict__ bias,
    const float* __restrict__ h0, const float* __restrict__ c0,
    f16* __restrict__ h16, float* __restrict__ c)
{
    int i = blockIdx.x * 256 + threadIdx.x;
    if (i < G4H) bias[i] = b_ih[i] + b_hh[i];
    if (i < B_ * H_) { h16[i] = (f16)h0[i]; c[i] = c0[i]; }
}

__global__ __launch_bounds__(256) void zero_f32_k(float* __restrict__ p, int n)
{
    int i = blockIdx.x * 256 + threadIdx.x;
    if (i < n) p[i] = 0.0f;
}

// ---------------------------------------------------------------------------
// LDS-free WMMA GEMM: out = act( A[M][lda] * Bw[n][ldb]^T + bias + addC ).
// A f16 row-major, Bw f16 row-major [Nfull][ldb] (torch W cast to f16,
// padded). Block tile 64x128, 8 waves: wave = (m-tile, n-group of 4 tiles),
// K chunks of 32. All fragments load directly from global (B is L2-resident
// and reused by every block). All 4 B fragments are issued before any wmma
// so load latency overlaps the XDL pipe instead of serializing per tile.
// ---------------------------------------------------------------------------
__global__ __launch_bounds__(256) void gemm_f16_k(
    const f16* __restrict__ A, int lda,
    const f16* __restrict__ Bw, int ldb,
    const float* __restrict__ bias,
    const float* __restrict__ addC, long addStride,
    float* __restrict__ C, f16* __restrict__ C16, int ldc,
    int Nreal, int Kpad, int relu)
{
    const int lane = threadIdx.x & 31, w = threadIdx.x >> 5;
    const int half = lane >> 4, ln = lane & 15;
    const int m0 = blockIdx.x * 64;
    const int n0 = blockIdx.y * 128;
    const int wm = w & 3;     // m-tile 0..3
    const int np = w >> 2;    // n group 0..1 (4 tiles each)
    const int mrow = m0 + wm * 16 + ln;

    const f16* Arow = A + (size_t)mrow * lda;
    const f16* Brow[4];
    #pragma unroll
    for (int nt = 0; nt < 4; ++nt) {
        int n = n0 + (np * 4 + nt) * 16 + ln;
        Brow[nt] = Bw + (size_t)n * ldb + half * 16;
    }

    v8f acc[4] = {};

    for (int kc = 0; kc < Kpad; kc += 32) {
        // A fragment: lane m fixed, k = 16*half + e contiguous (matches B)
        v8h alo = *(const v8h*)&Arow[kc + half * 16];
        v8h ahi = *(const v8h*)&Arow[kc + half * 16 + 8];
        // B fragments: lane n row fixed, k = 16*half + e contiguous.
        // Issue every load before the first wmma.
        v16h bf[4];
        #pragma unroll
        for (int nt = 0; nt < 4; ++nt) bf[nt] = *(const v16h*)&Brow[nt][kc];
        v16h a;
        #pragma unroll
        for (int e = 0; e < 8; ++e) { a[e] = alo[e]; a[e + 8] = ahi[e]; }
        if (kc + 32 < Kpad) __builtin_prefetch(&Arow[kc + 32], 0, 1);
        #pragma unroll
        for (int nt = 0; nt < 4; ++nt) {
            acc[nt] = __builtin_amdgcn_wmma_f32_16x16x32_f16(
                false, a, false, bf[nt], (short)0, acc[nt], false, false);
        }
    }

    // epilogue: lane holds (m = r + 8*half, n = ln) of each tile
    #pragma unroll
    for (int nt = 0; nt < 4; ++nt) {
        int n = n0 + (np * 4 + nt) * 16 + ln;
        if (n >= Nreal) continue;
        float bv = bias ? bias[n] : 0.0f;
        #pragma unroll
        for (int r = 0; r < 8; ++r) {
            int m = m0 + wm * 16 + r + half * 8;
            float v = acc[nt][r] + bv;
            if (addC) v += addC[(size_t)m * addStride + n];
            if (relu) v = fmaxf(v, 0.0f);
            if (C)   C[(size_t)m * ldc + n] = v;
            if (C16) C16[(size_t)m * ldc + n] = (f16)v;
        }
    }
}

// ---------------------------------------------------------------------------
// LSTM cell elementwise update (gate order i,f,g,o), captures h at lengths-1.
// ---------------------------------------------------------------------------
__device__ __forceinline__ float sigmoidf_(float x) { return 1.0f / (1.0f + __expf(-x)); }

__global__ __launch_bounds__(256) void lstm_cell_k(
    const float* __restrict__ gates, float* __restrict__ c,
    f16* __restrict__ h16, f16* __restrict__ q16,
    const int* __restrict__ lengths, int t)
{
    int idx = blockIdx.x * 256 + threadIdx.x;   // 0 .. B*H-1
    int b = idx >> 7, hc = idx & (H_ - 1);
    const float* g = gates + (size_t)b * G4H;
    float gi = g[hc], gf = g[H_ + hc], gg = g[2 * H_ + hc], go = g[3 * H_ + hc];
    float cn = sigmoidf_(gf) * c[idx] + sigmoidf_(gi) * tanhf(gg);
    float hn = sigmoidf_(go) * tanhf(cn);
    c[idx] = cn;
    h16[idx] = (f16)hn;
    if (lengths[b] - 1 == t) q16[idx] = (f16)hn;
}

// ---------------------------------------------------------------------------
// Fused relation g-network: block = 64 pair-rows x full 256 width.
// Activations double-buffered in 64 KB static LDS; weights stream straight
// from global (f16 [n][k] rows = B fragments in this code's k order). Two barriers per layer (one on the last).
// All 8 B fragments of a K-chunk are issued before any wmma. Final layer is
// ReLU-reduced over rows -> atomicAdd into x_g.
// ---------------------------------------------------------------------------
__global__ __launch_bounds__(256) void relation_fused_k(
    const float* __restrict__ state, const f16* __restrict__ q16,
    const f16* __restrict__ w0, const f16* __restrict__ w1,
    const f16* __restrict__ w2, const f16* __restrict__ w3,
    const float* __restrict__ b0, const float* __restrict__ b1,
    const float* __restrict__ b2, const float* __restrict__ b3,
    float* __restrict__ xg)
{
    __shared__ __attribute__((aligned(32))) f16 actA[64 * 256];
    __shared__ __attribute__((aligned(32))) f16 actB[64 * 256];

    const int tid  = threadIdx.x;
    const int lane = tid & 31, w = tid >> 5;
    const int half = lane >> 4, ln = lane & 15;
    const int b     = blockIdx.x >> 4;          // 16 blocks per batch element
    const int pair0 = (blockIdx.x & 15) * 64;

    // ---- build feature rows: [state[b,j,:4], state[b,i,:4], q[b,:128], pad] ----
    {
        int r = tid >> 2, part = tid & 3;       // 4 threads per row, 40 cols each
        int p = pair0 + r;
        int oi = p >> 5, oj = p & 31;
        for (int cc = part * 40; cc < part * 40 + 40; ++cc) {
            f16 v;
            if (cc < 4)        v = (f16)state[(size_t)b * O_ * 4 + oj * 4 + cc];
            else if (cc < 8)   v = (f16)state[(size_t)b * O_ * 4 + oi * 4 + (cc - 4)];
            else if (cc < 136) v = q16[(size_t)b * H_ + (cc - 8)];
            else               v = (f16)0.0f;
            actA[r * 256 + cc] = v;
        }
    }
    __syncthreads();

    const f16*   Wt[4]  = { w0, w1, w2, w3 };
    const float* Bs[4]  = { b0, b1, b2, b3 };
    const int    Ks[4]  = { 160, 256, 256, 256 };

    f16* cur = actA;
    f16* nxt = actB;
    const int mt = w >> 1;   // m-tile 0..3 (16 rows)
    const int nh = w & 1;    // n half (8 tiles = 128 cols)

    for (int layer = 0; layer < 4; ++layer) {
        v8f acc[8] = {};
        const int  K   = Ks[layer];
        const int  ldb = K;                       // weights are [256][K]
        const f16* Wl  = Wt[layer] + half * 16;
        for (int kc = 0; kc < K; kc += 32) {
            v8h alo = *(const v8h*)&cur[(mt * 16 + ln) * 256 + kc + half * 16];
            v8h ahi = *(const v8h*)&cur[(mt * 16 + ln) * 256 + kc + half * 16 + 8];
            // issue all 8 B-fragment loads up front
            v16h bf[8];
            #pragma unroll
            for (int nt = 0; nt < 8; ++nt) {
                int n = (nh * 8 + nt) * 16 + ln;
                bf[nt] = *(const v16h*)&Wl[(size_t)n * ldb + kc];
            }
            v16h a;
            #pragma unroll
            for (int e = 0; e < 8; ++e) { a[e] = alo[e]; a[e + 8] = ahi[e]; }
            #pragma unroll
            for (int nt = 0; nt < 8; ++nt) {
                acc[nt] = __builtin_amdgcn_wmma_f32_16x16x32_f16(
                    false, a, false, bf[nt], (short)0, acc[nt], false, false);
            }
        }
        __syncthreads();   // all reads of `cur` complete
        if (layer < 3) {
            const float* bl = Bs[layer];
            #pragma unroll
            for (int nt = 0; nt < 8; ++nt) {
                int n = (nh * 8 + nt) * 16 + ln;
                float bv = bl[n];
                #pragma unroll
                for (int r = 0; r < 8; ++r) {
                    float v = fmaxf(acc[nt][r] + bv, 0.0f);
                    nxt[(mt * 16 + r + half * 8) * 256 + n] = (f16)v;
                }
            }
            f16* tmp = cur; cur = nxt; nxt = tmp;
            __syncthreads();   // `nxt` writes visible before next layer reads
        } else {
            const float* bl = Bs[3];
            #pragma unroll
            for (int nt = 0; nt < 8; ++nt) {
                int n = (nh * 8 + nt) * 16 + ln;
                float bv = bl[n];
                float s = 0.0f;
                #pragma unroll
                for (int r = 0; r < 8; ++r) s += fmaxf(acc[nt][r] + bv, 0.0f);
                atomicAdd(&xg[(size_t)b * 256 + n], s);
            }
        }
    }
}

// ---------------------------------------------------------------------------
// Row-wise log_softmax over V=1000.
// ---------------------------------------------------------------------------
__global__ __launch_bounds__(256) void logsoftmax_k(
    const float* __restrict__ logits, float* __restrict__ out)
{
    __shared__ float red[256];
    int b = blockIdx.x, tid = threadIdx.x;
    const float* row = logits + (size_t)b * V_;
    float mx = -3.4e38f;
    for (int i = tid; i < V_; i += 256) mx = fmaxf(mx, row[i]);
    red[tid] = mx; __syncthreads();
    for (int s = 128; s > 0; s >>= 1) {
        if (tid < s) red[tid] = fmaxf(red[tid], red[tid + s]);
        __syncthreads();
    }
    float m = red[0]; __syncthreads();
    float sum = 0.0f;
    for (int i = tid; i < V_; i += 256) sum += __expf(row[i] - m);
    red[tid] = sum; __syncthreads();
    for (int s = 128; s > 0; s >>= 1) {
        if (tid < s) red[tid] += red[tid + s];
        __syncthreads();
    }
    float lse = __logf(red[0]) + m;
    for (int i = tid; i < V_; i += 256) out[(size_t)b * V_ + i] = row[i] - lse;
}

// ---------------------------------------------------------------------------
extern "C" void kernel_launch(void* const* d_in, const int* in_sizes, int n_in,
                              void* d_out, int out_size, void* d_ws, size_t ws_size,
                              hipStream_t stream) {
    (void)in_sizes; (void)n_in; (void)out_size; (void)ws_size;

    const float* ques   = (const float*)d_in[0];
    const float* state  = (const float*)d_in[1];
    const float* h0     = (const float*)d_in[2];
    const float* c0     = (const float*)d_in[3];
    const float* W_ih   = (const float*)d_in[4];
    const float* W_hh   = (const float*)d_in[5];
    const float* b_ih   = (const float*)d_in[6];
    const float* b_hh   = (const float*)d_in[7];
    const float* g1_w   = (const float*)d_in[8];
    const float* g1_b   = (const float*)d_in[9];
    const float* g2_w   = (const float*)d_in[10];
    const float* g2_b   = (const float*)d_in[11];
    const float* g3_w   = (const float*)d_in[12];
    const float* g3_b   = (const float*)d_in[13];
    const float* g4_w   = (const float*)d_in[14];
    const float* g4_b   = (const float*)d_in[15];
    const float* f1_w   = (const float*)d_in[16];
    const float* f1_b   = (const float*)d_in[17];
    const float* fc2_w  = (const float*)d_in[18];
    const float* fc2_b  = (const float*)d_in[19];
    const float* fc3_w  = (const float*)d_in[20];
    const float* fc3_b  = (const float*)d_in[21];
    const int*   lengths= (const int*)d_in[22];

    char* ws = (char*)d_ws;
    size_t off = 0;
    auto take = [&](size_t bytes) -> void* {
        void* p = ws + off;
        off = (off + bytes + 255) & ~(size_t)255;
        return p;
    };

    float* gates_pre = (float*)take((size_t)B_ * T_ * G4H * 4);     // 33.6 MB
    f16*   ques16    = (f16*)  take((size_t)B_ * T_ * 1024 * 2);    // 33.6 MB
    f16*   W16_ih    = (f16*)  take((size_t)G4H * 1024 * 2);
    f16*   W16_hh    = (f16*)  take((size_t)G4H * H_ * 2);
    f16*   W16_g1    = (f16*)  take((size_t)256 * 160 * 2);
    f16*   W16_g2    = (f16*)  take((size_t)256 * 256 * 2);
    f16*   W16_g3    = (f16*)  take((size_t)256 * 256 * 2);
    f16*   W16_g4    = (f16*)  take((size_t)256 * 256 * 2);
    f16*   W16_f1    = (f16*)  take((size_t)256 * 256 * 2);
    f16*   W16_fc2   = (f16*)  take((size_t)256 * 256 * 2);
    f16*   W16_fc3   = (f16*)  take((size_t)1024 * 256 * 2);
    float* bias_lstm = (float*)take((size_t)G4H * 4);
    f16*   h16       = (f16*)  take((size_t)B_ * H_ * 2);
    float* c_ws      = (float*)take((size_t)B_ * H_ * 4);
    f16*   q16       = (f16*)  take((size_t)B_ * H_ * 2);
    float* gates     = (float*)take((size_t)B_ * G4H * 4);
    float* xg        = (float*)take((size_t)B_ * 256 * 4);
    f16*   xg16      = (f16*)  take((size_t)B_ * 256 * 2);
    f16*   xf16      = (f16*)  take((size_t)B_ * 256 * 2);
    f16*   x2_16     = (f16*)  take((size_t)B_ * 256 * 2);
    float* logits    = (float*)take((size_t)B_ * V_ * 4);

    dim3 blk(256);
    auto cvtW = [&](const float* src, f16* dst, int N, int K, int Np, int Kp) {
        int elems = Np * Kp;
        cvt_weight16_k<<<dim3((elems + 255) / 256), blk, 0, stream>>>(src, dst, N, K, Np, Kp);
    };
    cvtW(W_ih,  W16_ih,  G4H, V_,  G4H,  1024);
    cvtW(W_hh,  W16_hh,  G4H, H_,  G4H,  H_);
    cvtW(g1_w,  W16_g1,  256, 136, 256,  160);
    cvtW(g2_w,  W16_g2,  256, 256, 256,  256);
    cvtW(g3_w,  W16_g3,  256, 256, 256,  256);
    cvtW(g4_w,  W16_g4,  256, 256, 256,  256);
    cvtW(f1_w,  W16_f1,  256, 256, 256,  256);
    cvtW(fc2_w, W16_fc2, 256, 256, 256,  256);
    cvtW(fc3_w, W16_fc3, V_,  256, 1024, 256);

    // ques f32 [16384][1000] -> f16 [16384][1024] (zero-padded K)
    {
        int elems = B_ * T_ * 1024;
        cvt_act16_k<<<dim3((elems + 255) / 256), blk, 0, stream>>>(
            ques, ques16, B_ * T_, V_, 1024);
    }

    init_lstm_k<<<dim3((B_ * H_ + 255) / 256), blk, 0, stream>>>(
        b_ih, b_hh, bias_lstm, h0, c0, h16, c_ws);

    // gates_pre = ques @ W_ih^T + (b_ih + b_hh)   [16384 x 512], K=1024
    gemm_f16_k<<<dim3(B_ * T_ / 64, G4H / 128), blk, 0, stream>>>(
        ques16, 1024, W16_ih, 1024, bias_lstm, nullptr, 0,
        gates_pre, nullptr, G4H, G4H, 1024, 0);

    // LSTM recurrence
    for (int t = 0; t < T_; ++t) {
        gemm_f16_k<<<dim3(B_ / 64, G4H / 128), blk, 0, stream>>>(
            h16, H_, W16_hh, H_, nullptr,
            gates_pre + (size_t)t * G4H, (long)(T_ * G4H),
            gates, nullptr, G4H, G4H, H_, 0);
        lstm_cell_k<<<dim3(B_ * H_ / 256), blk, 0, stream>>>(
            gates, c_ws, h16, q16, lengths, t);
    }

    // relation network (fused 4-layer MLP + sum over pairs)
    zero_f32_k<<<dim3((B_ * 256 + 255) / 256), blk, 0, stream>>>(xg, B_ * 256);
    relation_fused_k<<<dim3(B_ * 16), blk, 0, stream>>>(
        state, q16, W16_g1, W16_g2, W16_g3, W16_g4,
        g1_b, g2_b, g3_b, g4_b, xg);

    // head: xg -> f16, then f1(relu) -> fc2(relu) -> fc3 -> log_softmax
    cvt_act16_k<<<dim3((B_ * 256 + 255) / 256), blk, 0, stream>>>(xg, xg16, B_, 256, 256);
    gemm_f16_k<<<dim3(B_ / 64, 2), blk, 0, stream>>>(
        xg16, 256, W16_f1, 256, f1_b, nullptr, 0,
        nullptr, xf16, 256, 256, 256, 1);
    gemm_f16_k<<<dim3(B_ / 64, 2), blk, 0, stream>>>(
        xf16, 256, W16_fc2, 256, fc2_b, nullptr, 0,
        nullptr, x2_16, 256, 256, 256, 1);
    gemm_f16_k<<<dim3(B_ / 64, 8), blk, 0, stream>>>(
        x2_16, 256, W16_fc3, 256, fc3_b, nullptr, 0,
        logits, nullptr, V_, V_, 256, 0);

    logsoftmax_k<<<dim3(B_), blk, 0, stream>>>(logits, (float*)d_out);
}
